// URCE_84490596647386
// MI455X (gfx1250) — matrix operand order repaired and checked by reading, not compiled, not run on hardware
//
#include <hip/hip_runtime.h>
#include <hip/hip_bf16.h>

// ---------------------------------------------------------------------------
// CDNA5 bf16 WMMA building blocks (v_wmma_f32_16x16x32_bf16, wave32)
// ---------------------------------------------------------------------------
typedef __attribute__((ext_vector_type(16))) __bf16 v16bf;
typedef __attribute__((ext_vector_type(8)))  float  v8f;
typedef __attribute__((ext_vector_type(4)))  unsigned int v4u;
typedef __attribute__((ext_vector_type(4)))  int    v4i;
typedef __attribute__((ext_vector_type(8)))  int    v8i;

union FragAB { v16bf v; uint4 q[2]; };

__device__ __forceinline__ unsigned short f2bf(float x) {
    unsigned int u = __float_as_uint(x);
    unsigned int r = (u + 0x7FFFu + ((u >> 16) & 1u)) >> 16;
    return (unsigned short)r;
}
__device__ __forceinline__ float bf2f(unsigned short h) {
    return __uint_as_float(((unsigned int)h) << 16);
}
__device__ __forceinline__ v8f zero8() {
    v8f z;
    #pragma unroll
    for (int i = 0; i < 8; ++i) z[i] = 0.0f;
    return z;
}
// A fragment (16x32 bf16): lane holds row m=lane&15; K layout per ISA:
//   K = (e&7) + (e>=8 ? 16:0) + (lane>=16 ? 8:0)  -> two contiguous 8-half runs
__device__ __forceinline__ v16bf load_a(const unsigned short* base, int ldk, int lane) {
    const unsigned short* p = base + (lane & 15) * ldk + ((lane & 16) ? 8 : 0);
    FragAB f;
    f.q[0] = *(const uint4*)(p);
    f.q[1] = *(const uint4*)(p + 16);
    return f.v;
}
// B fragment (32x16 bf16): lane holds column n=lane&15, K = (e&7) + (e>=8 ? 16:0) + (lane>=16 ? 8:0)
// (LDS tile stored transposed [n][k], k contiguous)
__device__ __forceinline__ v16bf load_b(const unsigned short* base, int ldk, int lane) {
    const unsigned short* p = base + (lane & 15) * ldk + ((lane & 16) ? 8 : 0);
    FragAB f;
    f.q[0] = *(const uint4*)(p);
    f.q[1] = *(const uint4*)(p + 16);
    return f.v;
}
__device__ __forceinline__ v8f wmma_bf16(v16bf a, v16bf b, v8f c) {
    return __builtin_amdgcn_wmma_f32_16x16x32_bf16(false, a, false, b, (short)0, c,
                                                   false, false);
}

// ---------------------------------------------------------------------------
// Tensor Data Mover: 1-D bf16 tile, global -> LDS (D# per ISA 08 §8.3/8.4).
// Issue from one wave, then s_wait_tensorcnt + workgroup barrier publishes LDS.
// ---------------------------------------------------------------------------
__device__ __forceinline__ void tdm_load_1d(unsigned lds_byte_addr,
                                            const void* gptr, unsigned nelem) {
    unsigned long long ga = (unsigned long long)gptr;
    v4u g0;
    g0[0] = 1u;                                   // count=1, user descriptor
    g0[1] = lds_byte_addr;                        // lds_addr (bytes)
    g0[2] = (unsigned)(ga & 0xFFFFFFFFu);         // global_addr[31:0]  (bits 95:64)
    g0[3] = (unsigned)((ga >> 32) & 0x01FFFFFFu)  // global_addr[56:32] (bits 120:96)
          | (2u << 30);                           // type=2 ("image")
    v8i g1;
    #pragma unroll
    for (int i = 0; i < 8; ++i) g1[i] = 0;
    g1[0] = (int)(1u << 16);                      // data_size=1 -> 2 bytes/elem
    g1[1] = (int)((nelem & 0xFFFFu) << 16);       // tensor_dim0[15:0]  @ bits 63:48
    g1[2] = (int)((nelem >> 16) & 0xFFFFu);       // tensor_dim0[31:16] @ bits 79:64
    g1[3] = (int)((nelem & 0xFFFFu) << 16);       // tile_dim0 @ bits 127:112
    g1[5] = (int)nelem;                           // tensor_dim0_stride[31:0] @ 191:160
    v4i z4;
    #pragma unroll
    for (int i = 0; i < 4; ++i) z4[i] = 0;
#if defined(__clang_major__) && (__clang_major__ >= 23)
    v8i z8;
    #pragma unroll
    for (int i = 0; i < 8; ++i) z8[i] = 0;
    __builtin_amdgcn_tensor_load_to_lds(g0, g1, z4, z4, z8, 0);
#else
    __builtin_amdgcn_tensor_load_to_lds(g0, g1, z4, z4, 0);
#endif
}

// ---------------------------------------------------------------------------
// Kernel 1: saliency resize + |diff| map + top-32 per batch
// ---------------------------------------------------------------------------
__device__ __forceinline__ float bilin16(const float* img, int Hi, int oy, int ox) {
    float sc = (float)(Hi - 1) / 15.0f;
    float fy = oy * sc, fx = ox * sc;
    int y0 = (int)fy; int y1 = (y0 + 1 < Hi) ? y0 + 1 : Hi - 1;
    int x0 = (int)fx; int x1 = (x0 + 1 < Hi) ? x0 + 1 : Hi - 1;
    float wy = fy - (float)y0, wx = fx - (float)x0;
    float a = img[y0 * Hi + x0], b = img[y0 * Hi + x1];
    float c = img[y1 * Hi + x0], d = img[y1 * Hi + x1];
    float top = a * (1.0f - wx) + b * wx;
    float bot = c * (1.0f - wx) + d * wx;
    return top * (1.0f - wy) + bot * wy;
}

__global__ void k_topk(const float* __restrict__ s1, const float* __restrict__ s2,
                       const float* __restrict__ s3, const float* __restrict__ s4,
                       int* __restrict__ idx) {
    int b = blockIdx.x, t = threadIdx.x;
    __shared__ float mapU[256];
    __shared__ float rv[256];
    __shared__ int   ri[256];
    int oy = t >> 4, ox = t & 15;
    float r1 = bilin16(s1 + (size_t)b * 128 * 128, 128, oy, ox);
    float r2 = bilin16(s2 + (size_t)b * 64 * 64, 64, oy, ox);
    float r3 = bilin16(s3 + (size_t)b * 32 * 32, 32, oy, ox);
    float r4 = s4[(size_t)b * 256 + t];
    mapU[t] = fabsf(r1 - r2) + fabsf(r2 - r3) + fabsf(r3 - r4);
    __syncthreads();
    for (int kk = 0; kk < 32; ++kk) {
        rv[t] = mapU[t]; ri[t] = t;
        __syncthreads();
        for (int s = 128; s > 0; s >>= 1) {
            if (t < s) {
                if (rv[t + s] > rv[t] ||
                    (rv[t + s] == rv[t] && ri[t + s] < ri[t])) {
                    rv[t] = rv[t + s]; ri[t] = ri[t + s];
                }
            }
            __syncthreads();
        }
        if (t == 0) { idx[b * 32 + kk] = ri[0]; mapU[ri[0]] = -3.0e38f; }
        __syncthreads();
    }
}

// ---------------------------------------------------------------------------
// Kernel 2: fc projection (pixels x c) @ (c x 256), write 2*(val+bias) as bf16
// into token-rearranged xf. 256 threads = 8 waves; 32 pixels x 256 N per block.
// ---------------------------------------------------------------------------
__global__ void k_fc(const float* __restrict__ x, const float* __restrict__ w,
                     const float* __restrict__ bias, unsigned short* __restrict__ xf,
                     int c, int S, int win, int tokoff) {
    const int SS = S * S;
    const int bpb = SS / 32;
    int b = blockIdx.x / bpb;
    int pbase = (blockIdx.x % bpb) * 32;
    int t = threadIdx.x, lane = t & 31, wave = t >> 5;

    __shared__ __align__(16) unsigned short ldsA[32 * 32];
    __shared__ __align__(16) unsigned short ldsB[256 * 32];

    int mtile = wave & 1;
    int nq = wave >> 1;  // N base = nq*64
    v8f acc[4];
    #pragma unroll
    for (int f = 0; f < 4; ++f) acc[f] = zero8();

    for (int k0 = 0; k0 < c; k0 += 32) {
        __syncthreads();
        if (k0 + 32 < c)                        // prefetch next weight K-slab
            __builtin_prefetch(w + (size_t)(k0 + 32) * 256 + t, 0, 1);
        #pragma unroll
        for (int i = 0; i < 4; ++i) {          // A: 32 pixels x 32 K
            int e = t + 256 * i;
            int kl = e >> 5, m = e & 31;
            float v = x[((size_t)b * c + (k0 + kl)) * SS + pbase + m];
            ldsA[m * 32 + kl] = f2bf(v);
        }
        #pragma unroll 4
        for (int i = 0; i < 32; ++i)           // B transposed: [n][k]
            ldsB[t * 32 + i] = f2bf(w[(size_t)(k0 + i) * 256 + t]);
        __syncthreads();
        v16bf a = load_a(ldsA + mtile * 16 * 32, 32, lane);
        #pragma unroll
        for (int f = 0; f < 4; ++f) {
            v16bf bb = load_b(ldsB + (nq * 4 + f) * 16 * 32, 32, lane);
            acc[f] = wmma_bf16(a, bb, acc[f]);
        }
    }
    #pragma unroll
    for (int f = 0; f < 4; ++f) {
        int n = nq * 64 + f * 16 + (lane & 15);
        float bn = bias[n];
        #pragma unroll
        for (int r = 0; r < 8; ++r) {
            int m = r + ((lane >> 4) << 3) + mtile * 16;
            int p = pbase + m;
            int yy = p / S, xx = p % S;
            int cell = (yy / win) * 16 + (xx / win);
            int tok = tokoff + (yy % win) * win + (xx % win);
            size_t addr = (((size_t)b * 256 + cell) * 85 + tok) * 256 + n;
            xf[addr] = f2bf(2.0f * (acc[f][r] + bn));
        }
    }
}

// ---------------------------------------------------------------------------
// Kernel 3a: per selected group, y(96x256) @ qkv_w(256x768) -> q,k (row-major),
// v transposed. One block per group (256 groups). The y tile is DMA'd via the
// Tensor Data Mover (bf16 verbatim); the 0.5 un-scale folds into the epilogue.
// ---------------------------------------------------------------------------
__global__ void k_qkv(const unsigned short* __restrict__ xf, const int* __restrict__ idx,
                      const float* __restrict__ qkv_w, const float* __restrict__ qkv_b,
                      unsigned short* __restrict__ qws, unsigned short* __restrict__ kws,
                      unsigned short* __restrict__ vtws) {
    int g = blockIdx.x;
    int b = g >> 5, kk = g & 31;
    int cell = idx[b * 32 + kk];
    int t = threadIdx.x, lane = t & 31, wave = t >> 5;
    __shared__ __align__(16) unsigned short ldsY[96 * 256];
    __shared__ __align__(16) unsigned short ldsW[64 * 32];

    const unsigned short* src = xf + (((size_t)b * 256 + cell) * 85) * 256;
    for (int i = 85; i < 96; ++i)              // zero the pad rows
        ldsY[i * 256 + t] = 0;
    if (wave == 0) {                           // TDM: 85x256 bf16 tile -> LDS
        tdm_load_1d((unsigned)(size_t)(void*)ldsY, (const void*)src, 85 * 256);
        __builtin_amdgcn_s_wait_tensorcnt(0);
    }
    __syncthreads();

    size_t gbase = (size_t)g * 96 * 256;
    for (int nc = 0; nc < 12; ++nc) {          // 768 N in chunks of 64
        int nbase = nc * 64;
        v8f acc[3];
        #pragma unroll
        for (int j = 0; j < 3; ++j) acc[j] = zero8();
        for (int k0 = 0; k0 < 256; k0 += 32) {
            __syncthreads();
            #pragma unroll
            for (int i = 0; i < 8; ++i) {
                int e = t + 256 * i;
                int n = e & 63, kl = e >> 6;
                ldsW[n * 32 + kl] = f2bf(qkv_w[(size_t)(k0 + kl) * 768 + nbase + n]);
            }
            __syncthreads();
            #pragma unroll
            for (int j = 0; j < 3; ++j) {
                int tt = wave * 3 + j;          // 24 tiles = 6 M x 4 N
                int mt = tt % 6, nt = tt / 6;
                v16bf a = load_a(ldsY + mt * 16 * 256 + k0, 256, lane);
                v16bf bb = load_b(ldsW + nt * 16 * 32, 32, lane);
                acc[j] = wmma_bf16(a, bb, acc[j]);
            }
        }
        #pragma unroll
        for (int j = 0; j < 3; ++j) {
            int tt = wave * 3 + j;
            int mt = tt % 6, nt = tt / 6;
            int n = nbase + nt * 16 + (lane & 15);
            float bn = qkv_b[n];
            #pragma unroll
            for (int r = 0; r < 8; ++r) {
                int m = mt * 16 + r + ((lane >> 4) << 3);
                unsigned short h = f2bf(0.5f * acc[j][r] + bn);  // fold y=0.5*xf
                if (n < 256)      qws[gbase + (size_t)m * 256 + n] = h;
                else if (n < 512) kws[gbase + (size_t)m * 256 + (n - 256)] = h;
                else              vtws[gbase + (size_t)(n - 512) * 96 + m] = h;
            }
        }
    }
}

// ---------------------------------------------------------------------------
// Kernel 3b: per (group, head): scores = q k^T (K=32, one WMMA/tile),
// masked softmax, o = P @ v. 2048 blocks.
// ---------------------------------------------------------------------------
__global__ void k_attn(const unsigned short* __restrict__ qws,
                       const unsigned short* __restrict__ kws,
                       const unsigned short* __restrict__ vtws,
                       unsigned short* __restrict__ ows) {
    int g = blockIdx.x >> 3;
    int h = blockIdx.x & 7;
    int t = threadIdx.x, lane = t & 31, wave = t >> 5;
    __shared__ __align__(16) unsigned short ldsQ[96 * 32];
    __shared__ __align__(16) unsigned short ldsK[96 * 32];
    __shared__ __align__(16) unsigned short ldsVT[32 * 96];
    __shared__ __align__(16) unsigned short ldsP[96 * 96];

    size_t gq = (size_t)g * 96 * 256;
    #pragma unroll
    for (int i = 0; i < 12; ++i) {
        int e = t + 256 * i;
        int m = e >> 5, cc = e & 31;
        ldsQ[m * 32 + cc] = qws[gq + (size_t)m * 256 + h * 32 + cc];
        ldsK[m * 32 + cc] = kws[gq + (size_t)m * 256 + h * 32 + cc];
    }
    #pragma unroll
    for (int i = 0; i < 12; ++i) {
        int e = t + 256 * i;
        int d = e / 96, tok = e % 96;
        ldsVT[d * 96 + tok] = vtws[gq + (size_t)(h * 32 + d) * 96 + tok];
    }
    __syncthreads();
    for (int j = 0; j < 5; ++j) {              // 36 score tiles (6x6)
        int tt = wave + 8 * j;
        if (tt < 36) {
            int mt = tt % 6, nt = tt / 6;
            v8f acc = zero8();
            v16bf a = load_a(ldsQ + mt * 16 * 32, 32, lane);
            v16bf bb = load_b(ldsK + nt * 16 * 32, 32, lane);
            acc = wmma_bf16(a, bb, acc);
            int n = nt * 16 + (lane & 15);
            #pragma unroll
            for (int r = 0; r < 8; ++r) {
                int m = mt * 16 + r + ((lane >> 4) << 3);
                ldsP[m * 96 + n] = f2bf(acc[r]);
            }
        }
    }
    __syncthreads();
    if (t < 96) {                              // row softmax over 85 valid keys
        float mx = -3.0e38f;
        for (int n2 = 0; n2 < 85; ++n2) mx = fmaxf(mx, bf2f(ldsP[t * 96 + n2]));
        float sum = 0.0f;
        for (int n2 = 0; n2 < 85; ++n2) sum += __expf(bf2f(ldsP[t * 96 + n2]) - mx);
        float inv = 1.0f / sum;
        for (int n2 = 0; n2 < 85; ++n2)
            ldsP[t * 96 + n2] = f2bf(__expf(bf2f(ldsP[t * 96 + n2]) - mx) * inv);
        for (int n2 = 85; n2 < 96; ++n2) ldsP[t * 96 + n2] = 0;
    }
    __syncthreads();
    for (int j = 0; j < 2; ++j) {              // o tiles: 6 M x 2 N, K=96
        int tt = wave + 8 * j;
        if (tt < 12) {
            int mt = tt % 6, nt = tt / 6;
            v8f acc = zero8();
            #pragma unroll
            for (int ks = 0; ks < 3; ++ks) {
                v16bf a = load_a(ldsP + mt * 16 * 96 + ks * 32, 96, lane);
                v16bf bb = load_b(ldsVT + nt * 16 * 96 + ks * 32, 96, lane);
                acc = wmma_bf16(a, bb, acc);
            }
            int n = nt * 16 + (lane & 15);
            #pragma unroll
            for (int r = 0; r < 8; ++r) {
                int m = mt * 16 + r + ((lane >> 4) << 3);
                ows[gq + (size_t)m * 256 + h * 32 + n] = f2bf(acc[r]);
            }
        }
    }
}

// ---------------------------------------------------------------------------
// Kernel 3c: proj GEMM + layer_scale + residual, scatter back into xf.
// The o tile (96x256 bf16) is DMA'd via the Tensor Data Mover.
// ---------------------------------------------------------------------------
__global__ void k_proj(const unsigned short* __restrict__ ows, const int* __restrict__ idx,
                       const float* __restrict__ proj_w, const float* __restrict__ proj_b,
                       const float* __restrict__ ls, unsigned short* __restrict__ xf) {
    int g = blockIdx.x;
    int b = g >> 5, kk = g & 31;
    int cell = idx[b * 32 + kk];
    int t = threadIdx.x, lane = t & 31, wave = t >> 5;
    __shared__ __align__(16) unsigned short ldsO[96 * 256];
    __shared__ __align__(16) unsigned short ldsW[64 * 32];

    size_t gbase = (size_t)g * 96 * 256;
    if (wave == 0) {                           // TDM: 96x256 bf16 tile -> LDS
        tdm_load_1d((unsigned)(size_t)(void*)ldsO, (const void*)(ows + gbase), 96 * 256);
        __builtin_amdgcn_s_wait_tensorcnt(0);
    }
    __syncthreads();

    unsigned short* dst = xf + (((size_t)b * 256 + cell) * 85) * 256;
    for (int nc = 0; nc < 4; ++nc) {
        int nbase = nc * 64;
        v8f acc[3];
        #pragma unroll
        for (int j = 0; j < 3; ++j) acc[j] = zero8();
        for (int k0 = 0; k0 < 256; k0 += 32) {
            __syncthreads();
            #pragma unroll
            for (int i = 0; i < 8; ++i) {
                int e = t + 256 * i;
                int n = e & 63, kl = e >> 6;
                ldsW[n * 32 + kl] = f2bf(proj_w[(size_t)(k0 + kl) * 256 + nbase + n]);
            }
            __syncthreads();
            #pragma unroll
            for (int j = 0; j < 3; ++j) {
                int tt = wave * 3 + j;
                int mt = tt % 6, nt = tt / 6;
                v16bf a = load_a(ldsO + mt * 16 * 256 + k0, 256, lane);
                v16bf bb = load_b(ldsW + nt * 16 * 32, 32, lane);
                acc[j] = wmma_bf16(a, bb, acc[j]);
            }
        }
        #pragma unroll
        for (int j = 0; j < 3; ++j) {
            int tt = wave * 3 + j;
            int mt = tt % 6, nt = tt / 6;
            int n = nbase + nt * 16 + (lane & 15);
            float bn = proj_b[n], sn = ls[n];
            #pragma unroll
            for (int r = 0; r < 8; ++r) {
                int m = mt * 16 + r + ((lane >> 4) << 3);
                if (m < 85) {
                    size_t addr = (size_t)m * 256 + n;
                    float y0 = bf2f(dst[addr]) * 0.5f;   // undo 2x scale
                    dst[addr] = f2bf(y0 + sn * (acc[j][r] + bn));
                }
            }
        }
    }
}

// ---------------------------------------------------------------------------
// Kernel 4: fr projection (pixels x 256) @ (256 x c) -> fp32 outputs.
// ---------------------------------------------------------------------------
__global__ void k_fr(const unsigned short* __restrict__ xf, const float* __restrict__ w,
                     const float* __restrict__ bias, float* __restrict__ out,
                     int c, int S, int win, int tokoff) {
    const int SS = S * S;
    const int bpb = SS / 32;
    int b = blockIdx.x / bpb;
    int pbase = (blockIdx.x % bpb) * 32;
    int t = threadIdx.x, lane = t & 31, wave = t >> 5;
    __shared__ __align__(16) unsigned short ldsA[32 * 256];
    __shared__ __align__(16) unsigned short ldsW[64 * 32];

    #pragma unroll 4
    for (int i = 0; i < 32; ++i) {             // gather 32 token rows
        int p = pbase + i;
        int yy = p / S, xx = p % S;
        int cell = (yy / win) * 16 + (xx / win);
        int tok = tokoff + (yy % win) * win + (xx % win);
        ldsA[i * 256 + t] = xf[(((size_t)b * 256 + cell) * 85 + tok) * 256 + t];
    }
    __syncthreads();

    int mtile = wave & 1, ntile = wave >> 1;   // 2 M x 4 N tiles, one per wave
    int nchunks = c / 64;
    for (int nc = 0; nc < nchunks; ++nc) {
        int nbase = nc * 64;
        v8f acc = zero8();
        for (int k0 = 0; k0 < 256; k0 += 32) {
            __syncthreads();
            if (k0 + 32 < 256)                 // prefetch next weight K-slab
                __builtin_prefetch(w + (size_t)(k0 + 32) * c + nbase + (t & 63), 0, 1);
            #pragma unroll
            for (int i = 0; i < 8; ++i) {
                int e = t + 256 * i;
                int n = e & 63, kl = e >> 6;
                ldsW[n * 32 + kl] = f2bf(w[(size_t)(k0 + kl) * c + nbase + n]);
            }
            __syncthreads();
            v16bf a = load_a(ldsA + mtile * 16 * 256 + k0, 256, lane);
            v16bf bb = load_b(ldsW + ntile * 16 * 32, 32, lane);
            acc = wmma_bf16(a, bb, acc);
        }
        int ch = nbase + ntile * 16 + (lane & 15);
        float bn = bias[ch];
        #pragma unroll
        for (int r = 0; r < 8; ++r) {
            int m = mtile * 16 + r + ((lane >> 4) << 3);
            out[((size_t)b * c + ch) * SS + pbase + m] = acc[r] + bn;
        }
    }
}

// ---------------------------------------------------------------------------
// Launch
// ---------------------------------------------------------------------------
extern "C" void kernel_launch(void* const* d_in, const int* in_sizes, int n_in,
                              void* d_out, int out_size, void* d_ws, size_t ws_size,
                              hipStream_t stream) {
    (void)in_sizes; (void)n_in; (void)out_size; (void)ws_size;
    // dict order: per level i: x, s, fc_w, fc_b, fr_w, fr_b ; then qkv/proj/ls
    const float* x[4];   const float* s[4];
    const float* fcw[4]; const float* fcb[4];
    const float* frw[4]; const float* frb[4];
    for (int j = 0; j < 4; ++j) {
        x[j]   = (const float*)d_in[6 * j + 0];
        s[j]   = (const float*)d_in[6 * j + 1];
        fcw[j] = (const float*)d_in[6 * j + 2];
        fcb[j] = (const float*)d_in[6 * j + 3];
        frw[j] = (const float*)d_in[6 * j + 4];
        frb[j] = (const float*)d_in[6 * j + 5];
    }
    const float* qkv_w  = (const float*)d_in[24];
    const float* qkv_b  = (const float*)d_in[25];
    const float* proj_w = (const float*)d_in[26];
    const float* proj_b = (const float*)d_in[27];
    const float* ls     = (const float*)d_in[28];

    char* ws = (char*)d_ws;
    int* idx = (int*)ws;
    unsigned short* xf = (unsigned short*)(ws + 4096);
    const size_t xf_elems = (size_t)8 * 256 * 85 * 256;
    const size_t g_elems  = (size_t)256 * 96 * 256;
    unsigned short* qws  = xf + xf_elems;
    unsigned short* kws  = qws + g_elems;
    unsigned short* vtws = kws + g_elems;
    unsigned short* ows  = vtws + g_elems;

    const int S[4]    = {128, 64, 32, 16};
    const int C[4]    = {64, 128, 320, 512};
    const int WIN[4]  = {8, 4, 2, 1};
    const int TOFF[4] = {0, 64, 80, 84};

    k_topk<<<8, 256, 0, stream>>>(s[0], s[1], s[2], s[3], idx);
    for (int j = 0; j < 4; ++j)
        k_fc<<<8 * S[j] * S[j] / 32, 256, 0, stream>>>(x[j], fcw[j], fcb[j], xf,
                                                       C[j], S[j], WIN[j], TOFF[j]);
    k_qkv<<<256, 256, 0, stream>>>(xf, idx, qkv_w, qkv_b, qws, kws, vtws);
    k_attn<<<2048, 256, 0, stream>>>(qws, kws, vtws, ows);
    k_proj<<<256, 256, 0, stream>>>(ows, idx, proj_w, proj_b, ls, xf);

    float* out = (float*)d_out;
    size_t ooff = 0;
    for (int j = 0; j < 4; ++j) {
        k_fr<<<8 * S[j] * S[j] / 32, 256, 0, stream>>>(xf, frw[j], frb[j], out + ooff,
                                                       C[j], S[j], WIN[j], TOFF[j]);
        ooff += (size_t)8 * C[j] * S[j] * S[j];
    }
}
